// KQEnergyBlock_22874995819082
// MI455X (gfx1250) — matrix operand order repaired and checked by reading, not compiled, not run on hardware
//
#include <hip/hip_runtime.h>
#include <hip/hip_bf16.h>

// ---------------------------------------------------------------------------
// KQ-aligned attention + Hopfield MLP, restructured for CDNA5 WMMA (gfx1250).
//   T1 = (softmax(b*QK^T) @ K) @ Wq      (flash pass 1)
//   T2 = (softmax(b*QK^T)^T @ Q) @ Wk    (pass 2, reuses row stats m,l)
//   out = T1 + T2 + relu(x W_mlp^T) W_mlp
// Matmuls on v_wmma_f32_16x16x32_bf16; GEMM tiles staged via the Tensor Data
// Mover (tensor_load_to_lds + s_wait_tensorcnt); attention staging prefetched
// with global_prefetch_b8.
// ---------------------------------------------------------------------------

typedef __attribute__((ext_vector_type(16))) __bf16 v16bf;
typedef __attribute__((ext_vector_type(8)))  float  v8f;
typedef unsigned int u32x4 __attribute__((ext_vector_type(4)));
typedef int          i32x4 __attribute__((ext_vector_type(4)));
typedef int          i32x8 __attribute__((ext_vector_type(8)));

#define Bc 8
#define Nc 1024
#define Dc 768
#define Hc 12
#define Zc 64
#define HIDc 3072
#define Mrows (Bc * Nc)     // 8192

#define SA 40               // LDS stride for K=32 tiles (32 + 8 pad), mult. of 8
#define SZ 72               // LDS stride for 64-wide tiles (64 + 8 pad), mult. of 8

__device__ __forceinline__ unsigned short f2bf(float f) {
  union { float f; unsigned u; } v; v.f = f;
  unsigned r = v.u + 0x7FFFu + ((v.u >> 16) & 1u);   // round-to-nearest-even
  return (unsigned short)(r >> 16);
}

// ---------------------------------------------------------------------------
// Tensor Data Mover: DMA a 2D bf16 tile (rows x 32) from a row-major tensor
// (row stride kd elements) into LDS with an 8-element pad after each 32-elem
// row (LDS row stride = SA = 40 elements = 80 bytes).
//   D# group0: count=1 | lds_addr | global_addr(57b) | type=2
//   D# group1: data_size=2B, pad_enable, pad_interval=16 DW, pad_amount=4 DW,
//              tensor_dim0=kd, tensor_dim1=trows, tile_dim0=32, tile_dim1=rows,
//              tensor_dim0_stride=kd
// Issued by one wave; EXEC is ignored by TDM. Tracked by TENSORcnt.
// ---------------------------------------------------------------------------
__device__ __forceinline__ void tdm_load_tile32(const unsigned short* gptr,
                                                unsigned lds_addr,
                                                int kd, int rows, int trows) {
  unsigned long long ga = (unsigned long long)(uintptr_t)gptr;
  u32x4 g0;
  g0[0] = 1u;                                              // count = 1
  g0[1] = lds_addr;                                        // LDS byte address
  g0[2] = (unsigned)ga;                                    // global_addr[31:0]
  g0[3] = (unsigned)((ga >> 32) & 0x01FFFFFFull) | (2u << 30); // addr[56:32]|type=2
  i32x8 g1;
  g1[0] = (1 << 16) | (1 << 20) | (3 << 22) | (3 << 25);   // 2B, pad 4DW/16DW
  g1[1] = (kd & 0xFFFF) << 16;                             // tensor_dim0[15:0]
  g1[2] = ((kd >> 16) & 0xFFFF) | ((trows & 0xFFFF) << 16);// dim0 hi | dim1 lo
  g1[3] = ((trows >> 16) & 0xFFFF) | (32 << 16);           // dim1 hi | tile_dim0
  g1[4] = rows & 0xFFFF;                                   // tile_dim1 (tile_dim2=0)
  g1[5] = kd;                                              // tensor_dim0_stride lo
  g1[6] = 0;                                               // stride hi | dim1_stride lo
  g1[7] = 0;
  i32x4 z4 = {0, 0, 0, 0};
#if defined(__clang_major__) && __clang_major__ >= 23
  i32x8 z8 = {0, 0, 0, 0, 0, 0, 0, 0};
  __builtin_amdgcn_tensor_load_to_lds(g0, g1, z4, z4, z8, 0);
#else
  __builtin_amdgcn_tensor_load_to_lds(g0, g1, z4, z4, 0);
#endif
}

// A fragment (16x32, M x K), tile row-major [m][k] in LDS.
__device__ __forceinline__ v16bf frag_a16(const unsigned short* tile, int stride,
                                          int row0, int koff, int lane) {
  const int m = lane & 15, hi = lane >> 4;
  const unsigned short* p = tile + (row0 + m) * stride + koff + 8 * hi;
  union { v16bf v; uint4 q[2]; } f;
  f.q[0] = *(const uint4*)(p);
  f.q[1] = *(const uint4*)(p + 16);
  return f.v;
}

// B fragment (32x16, K x N), tile stored TRANSPOSED in LDS as [n][k].
__device__ __forceinline__ v16bf frag_b16(const unsigned short* tileT, int stride,
                                          int n0, int koff, int lane) {
  const int n = lane & 15, hi = lane >> 4;
  const unsigned short* p = tileT + (n0 + n) * stride + koff + 8 * hi;
  union { v16bf v; uint4 q[2]; } f;
  f.q[0] = *(const uint4*)(p);
  f.q[1] = *(const uint4*)(p + 16);
  return f.v;
}

__device__ __forceinline__ v8f wmma_bf16(v16bf a, v16bf b, v8f c) {
  return __builtin_amdgcn_wmma_f32_16x16x32_bf16(false, a, false, b,
                                                 (short)0, c, false, false);
}

// ---------------------------------------------------------------------------
// f32 -> bf16 conversion (vectorized x4)
// ---------------------------------------------------------------------------
__global__ __launch_bounds__(256)
void cvt_bf16_kernel(const float* __restrict__ in, unsigned short* __restrict__ out,
                     int n4) {
  int i = blockIdx.x * blockDim.x + threadIdx.x;
  if (i < n4) {
    float4 f = ((const float4*)in)[i];
    ushort4 o;
    o.x = f2bf(f.x); o.y = f2bf(f.y); o.z = f2bf(f.z); o.w = f2bf(f.w);
    ((ushort4*)out)[i] = o;
  }
}

// ---------------------------------------------------------------------------
// Generic bf16 GEMM: C[M,N] = A[M,K] * B, tile 128x64, 8 waves of 32x32.
// A tile (and B tile when BT) staged by the Tensor Data Mover.
// BT=true : B stored row-major [n][k] ; BT=false: B stored row-major [k][n].
// EPI: 0 = Cf  = v ; 1 = Cf += v ; 2 = Cb = bf16(relu(v)) ; 3 = Cb = bf16(v)
// ---------------------------------------------------------------------------
template<int EPI, bool BT>
__global__ __launch_bounds__(256)
void gemm_bf16_kernel(const unsigned short* __restrict__ A,
                      const unsigned short* __restrict__ B,
                      float* __restrict__ Cf, unsigned short* __restrict__ Cb,
                      int M, int Nn, int Kd) {
  __shared__ unsigned short sA[128 * SA];
  __shared__ unsigned short sB[64 * SA];      // [n][k]
  const int tid  = threadIdx.x;
  const int lane = tid & 31;
  const int wave = tid >> 5;
  const int bm = blockIdx.x * 128;
  const int bn = blockIdx.y * 64;
  const int wm = (wave & 3) * 32;
  const int wn = (wave >> 2) * 32;
  const unsigned ldsA = (unsigned)(uintptr_t)(void*)sA;
  const unsigned ldsB = (unsigned)(uintptr_t)(void*)sB;

  v8f acc[2][2] = {};
  const int nK = Kd >> 5;
  for (int ks = 0; ks < nK; ++ks) {
    const int k0 = ks * 32;
    if (wave == 0) {
      tdm_load_tile32(&A[(size_t)bm * Kd + k0], ldsA, Kd, 128, M);
      if (BT) tdm_load_tile32(&B[(size_t)bn * Kd + k0], ldsB, Kd, 64, Nn);
      __builtin_amdgcn_s_wait_tensorcnt(0);
    }
    if (!BT) {
      // B is [k][n]; transpose into sB[n][k]
      int kk = tid & 31, n0 = (tid >> 5) * 8;
      uint4 vv = *(const uint4*)(&B[(size_t)(k0 + kk) * Nn + bn + n0]);
      if (ks + 1 < nK)
        __builtin_prefetch(&B[(size_t)(k0 + 32 + kk) * Nn + bn + n0], 0, 0);
      const unsigned short* pv = (const unsigned short*)&vv;
      for (int j = 0; j < 8; ++j) sB[(n0 + j) * SA + kk] = pv[j];
    }
    __syncthreads();

    v16bf a0 = frag_a16(sA, SA, wm,      0, lane);
    v16bf a1 = frag_a16(sA, SA, wm + 16, 0, lane);
    v16bf b0 = frag_b16(sB, SA, wn,      0, lane);
    v16bf b1 = frag_b16(sB, SA, wn + 16, 0, lane);
    acc[0][0] = wmma_bf16(a0, b0, acc[0][0]);
    acc[0][1] = wmma_bf16(a0, b1, acc[0][1]);
    acc[1][0] = wmma_bf16(a1, b0, acc[1][0]);
    acc[1][1] = wmma_bf16(a1, b1, acc[1][1]);
    __syncthreads();
  }

  const int n = lane & 15, hi = lane >> 4;
  for (int i = 0; i < 2; ++i)
    for (int j2 = 0; j2 < 2; ++j2) {
      int col = bn + wn + j2 * 16 + n;
      for (int e = 0; e < 8; ++e) {
        int row = bm + wm + i * 16 + e + 8 * hi;
        float v = acc[i][j2][e];
        size_t idx = (size_t)row * Nn + col;
        if      (EPI == 0) Cf[idx]  = v;
        else if (EPI == 1) Cf[idx] += v;
        else if (EPI == 2) Cb[idx]  = f2bf(v > 0.f ? v : 0.f);
        else               Cb[idx]  = f2bf(v);
      }
    }
}

// ---------------------------------------------------------------------------
// Flash pass 1: per (b,h, 64-q tile), online softmax over k; writes
// AV = softmax(beta*QK^T) @ K  (bf16, laid out [b,n, h*64+z]), plus m,l.
// Block = 128 threads = 4 waves; wave w owns q rows [w*16, w*16+16).
// ---------------------------------------------------------------------------
__global__ __launch_bounds__(128)
void attn_pass1_kernel(const unsigned short* __restrict__ Qb,
                       const unsigned short* __restrict__ Kb,
                       const float* __restrict__ betas,
                       unsigned short* __restrict__ AVb,
                       float* __restrict__ mbuf, float* __restrict__ lbuf) {
  __shared__ unsigned short sQ[64 * SZ];      // [q][z]
  __shared__ unsigned short sK[64 * SZ];      // [k][z]  (= B^T tile for S-gemm)
  __shared__ unsigned short sKT[64 * SZ];     // [z][k]  (= B^T tile for AV-gemm)
  __shared__ unsigned short sP[4][16 * SZ];   // wave-private P tile [q16][k64]
  const int tid = threadIdx.x, lane = tid & 31, wave = tid >> 5;
  const int bh = blockIdx.y, b = bh / Hc, h = bh % Hc;
  const int q0 = blockIdx.x * 64;
  const float beta = betas[h];
  const int n = lane & 15, hi = lane >> 4;

  for (int c = tid; c < 512; c += 128) {
    int r = c >> 3, cc = (c & 7) * 8;
    *(uint4*)(&sQ[r * SZ + cc]) =
        *(const uint4*)(&Qb[(size_t)(b * Nc + q0 + r) * Dc + h * Zc + cc]);
  }

  float mcur[8], lcur[8];
  for (int j = 0; j < 8; ++j) { mcur[j] = -1e30f; lcur[j] = 0.f; }
  v8f acc[4] = {};

  for (int kb = 0; kb < Nc / 64; ++kb) {
    const int k0 = kb * 64;
    for (int c = tid; c < 512; c += 128) {
      int r = c >> 3, cc = (c & 7) * 8;
      uint4 vv = *(const uint4*)(&Kb[(size_t)(b * Nc + k0 + r) * Dc + h * Zc + cc]);
      if (kb + 1 < Nc / 64)   // prefetch next k-block into GL2
        __builtin_prefetch(&Kb[(size_t)(b * Nc + k0 + 64 + r) * Dc + h * Zc + cc], 0, 0);
      *(uint4*)(&sK[r * SZ + cc]) = vv;
      const unsigned short* pv = (const unsigned short*)&vv;
      for (int j = 0; j < 8; ++j) sKT[(cc + j) * SZ + r] = pv[j];
    }
    __syncthreads();

    // S (16q x 64k) = Q_tile @ K_tile^T
    v8f s[4] = {};
    for (int zz = 0; zz < 64; zz += 32) {
      v16bf aq = frag_a16(sQ, SZ, wave * 16, zz, lane);
      for (int t = 0; t < 4; ++t) {
        v16bf bk = frag_b16(sK, SZ, t * 16, zz, lane);
        s[t] = wmma_bf16(aq, bk, s[t]);
      }
    }

    // online softmax (rows m = e + 8*hi, cols across lanes 0..15 of each half)
    float mx[8];
    for (int j = 0; j < 8; ++j) {
      float v = s[0][j];
      for (int t = 1; t < 4; ++t) v = fmaxf(v, s[t][j]);
      mx[j] = v * beta;
    }
    for (int mask = 1; mask <= 8; mask <<= 1)
      for (int j = 0; j < 8; ++j) mx[j] = fmaxf(mx[j], __shfl_xor(mx[j], mask, 32));

    float alpha[8], rsum[8];
    for (int j = 0; j < 8; ++j) {
      float nm = fmaxf(mcur[j], mx[j]);
      alpha[j] = __expf(mcur[j] - nm);
      mcur[j] = nm;
      rsum[j] = 0.f;
    }
    unsigned short* P = sP[wave];
    for (int t = 0; t < 4; ++t)
      for (int j = 0; j < 8; ++j) {
        float p = __expf(s[t][j] * beta - mcur[j]);
        rsum[j] += p;
        P[(j + 8 * hi) * SZ + t * 16 + n] = f2bf(p);   // [q-row][k-col]
      }
    for (int mask = 1; mask <= 8; mask <<= 1)
      for (int j = 0; j < 8; ++j) rsum[j] += __shfl_xor(rsum[j], mask, 32);
    for (int j = 0; j < 8; ++j) lcur[j] = lcur[j] * alpha[j] + rsum[j];
    for (int t = 0; t < 4; ++t)
      for (int j = 0; j < 8; ++j) acc[t][j] *= alpha[j];

    asm volatile("" ::: "memory");   // order wave-private LDS write -> read
    // AV += P(16x64) @ K_tile(64x64); B^T layout = sKT [z][k]
    for (int kk = 0; kk < 64; kk += 32) {
      v16bf ap = frag_a16(P, SZ, 0, kk, lane);
      for (int t = 0; t < 4; ++t) {
        v16bf bk = frag_b16(sKT, SZ, t * 16, kk, lane);
        acc[t] = wmma_bf16(ap, bk, acc[t]);
      }
    }
    __syncthreads();
  }

  for (int t = 0; t < 4; ++t)
    for (int j = 0; j < 8; ++j) {
      int row = q0 + wave * 16 + j + 8 * hi;
      float av = acc[t][j] / lcur[j];
      AVb[(size_t)(b * Nc + row) * Dc + h * Zc + t * 16 + n] = f2bf(av);
    }
  if (n == 0)
    for (int j = 0; j < 8; ++j) {
      int row = q0 + wave * 16 + j + 8 * hi;
      mbuf[(size_t)bh * Nc + row] = mcur[j];
      lbuf[(size_t)bh * Nc + row] = lcur[j];
    }
}

// ---------------------------------------------------------------------------
// Pass 2: per (b,h, 64-k tile) recompute S^T = K Q^T, normalize with saved
// (m_q,l_q), accumulate AQ = attn^T @ Q (bf16, [b,n, h*64+z]).
// ---------------------------------------------------------------------------
__global__ __launch_bounds__(128)
void attn_pass2_kernel(const unsigned short* __restrict__ Qb,
                       const unsigned short* __restrict__ Kb,
                       const float* __restrict__ betas,
                       const float* __restrict__ mbuf,
                       const float* __restrict__ lbuf,
                       unsigned short* __restrict__ AQb) {
  __shared__ unsigned short sK[64 * SZ];      // [k][z]
  __shared__ unsigned short sQ[64 * SZ];      // [q][z]  (= B^T tile for S^T-gemm)
  __shared__ unsigned short sQT[64 * SZ];     // [z][q]  (= B^T tile for AQ-gemm)
  __shared__ unsigned short sP[4][16 * SZ];   // [k16][q64]
  const int tid = threadIdx.x, lane = tid & 31, wave = tid >> 5;
  const int bh = blockIdx.y, b = bh / Hc, h = bh % Hc;
  const int k0 = blockIdx.x * 64;
  const float beta = betas[h];
  const int n = lane & 15, hi = lane >> 4;

  for (int c = tid; c < 512; c += 128) {
    int r = c >> 3, cc = (c & 7) * 8;
    *(uint4*)(&sK[r * SZ + cc]) =
        *(const uint4*)(&Kb[(size_t)(b * Nc + k0 + r) * Dc + h * Zc + cc]);
  }

  v8f acc[4] = {};
  for (int qb2 = 0; qb2 < Nc / 64; ++qb2) {
    const int q0 = qb2 * 64;
    __syncthreads();                 // previous iter's frag reads done
    for (int c = tid; c < 512; c += 128) {
      int r = c >> 3, cc = (c & 7) * 8;
      uint4 vv = *(const uint4*)(&Qb[(size_t)(b * Nc + q0 + r) * Dc + h * Zc + cc]);
      if (qb2 + 1 < Nc / 64)
        __builtin_prefetch(&Qb[(size_t)(b * Nc + q0 + 64 + r) * Dc + h * Zc + cc], 0, 0);
      *(uint4*)(&sQ[r * SZ + cc]) = vv;
      const unsigned short* pv = (const unsigned short*)&vv;
      for (int j = 0; j < 8; ++j) sQT[(cc + j) * SZ + r] = pv[j];
    }
    __syncthreads();

    // S^T (16k x 64q) = K_tile @ Q_tile^T
    v8f s[4] = {};
    for (int zz = 0; zz < 64; zz += 32) {
      v16bf ak = frag_a16(sK, SZ, wave * 16, zz, lane);
      for (int t = 0; t < 4; ++t) {
        v16bf bq = frag_b16(sQ, SZ, t * 16, zz, lane);
        s[t] = wmma_bf16(ak, bq, s[t]);
      }
    }

    unsigned short* P = sP[wave];
    for (int t = 0; t < 4; ++t) {
      int qcol = q0 + t * 16 + n;              // column index = q
      float mq  = mbuf[(size_t)bh * Nc + qcol];
      float inv = 1.f / lbuf[(size_t)bh * Nc + qcol];
      for (int j = 0; j < 8; ++j) {
        float p = __expf(s[t][j] * beta - mq) * inv;
        P[(j + 8 * hi) * SZ + t * 16 + n] = f2bf(p);   // [k-row][q-col]
      }
    }
    asm volatile("" ::: "memory");

    // AQ += P^T-tile(16k x 64q) @ Q_tile(64q x 64z); B^T layout = sQT [z][q]
    for (int kk = 0; kk < 64; kk += 32) {
      v16bf ap = frag_a16(P, SZ, 0, kk, lane);
      for (int t = 0; t < 4; ++t) {
        v16bf bq = frag_b16(sQT, SZ, t * 16, kk, lane);
        acc[t] = wmma_bf16(ap, bq, acc[t]);
      }
    }
  }

  for (int t = 0; t < 4; ++t)
    for (int j = 0; j < 8; ++j) {
      int row = k0 + wave * 16 + j + 8 * hi;
      AQb[(size_t)(b * Nc + row) * Dc + h * Zc + t * 16 + n] = f2bf(acc[t][j]);
    }
}

// ---------------------------------------------------------------------------
extern "C" void kernel_launch(void* const* d_in, const int* in_sizes, int n_in,
                              void* d_out, int out_size, void* d_ws, size_t ws_size,
                              hipStream_t stream) {
  (void)in_sizes; (void)n_in; (void)out_size; (void)ws_size;
  const float* x     = (const float*)d_in[0];
  const float* Wq    = (const float*)d_in[1];
  const float* Wk    = (const float*)d_in[2];
  const float* betas = (const float*)d_in[3];
  const float* Wmlp  = (const float*)d_in[4];
  float* out = (float*)d_out;

  char* ws = (char*)d_ws;
  auto alloc = [&](size_t bytes) {
    char* p = ws; ws += (bytes + 255) & ~(size_t)255; return p;
  };
  unsigned short* xh   = (unsigned short*)alloc((size_t)Mrows * Dc * 2);
  unsigned short* Wqh  = (unsigned short*)alloc((size_t)Dc * Dc * 2);
  unsigned short* Wkh  = (unsigned short*)alloc((size_t)Dc * Dc * 2);
  unsigned short* Wmh  = (unsigned short*)alloc((size_t)HIDc * Dc * 2);
  unsigned short* Qh   = (unsigned short*)alloc((size_t)Mrows * Dc * 2);
  unsigned short* Kh   = (unsigned short*)alloc((size_t)Mrows * Dc * 2);
  unsigned short* AVb  = (unsigned short*)alloc((size_t)Mrows * Dc * 2);
  unsigned short* AQb  = (unsigned short*)alloc((size_t)Mrows * Dc * 2);
  unsigned short* hidb = (unsigned short*)alloc((size_t)Mrows * HIDc * 2);
  float* mbuf = (float*)alloc((size_t)Bc * Hc * Nc * 4);
  float* lbuf = (float*)alloc((size_t)Bc * Hc * Nc * 4);

  // 1) f32 -> bf16
  auto cvt = [&](const float* src, unsigned short* dst, size_t nelem) {
    int n4 = (int)(nelem / 4);
    cvt_bf16_kernel<<<(n4 + 255) / 256, 256, 0, stream>>>(src, dst, n4);
  };
  cvt(x,    xh,  (size_t)Mrows * Dc);
  cvt(Wq,   Wqh, (size_t)Dc * Dc);
  cvt(Wk,   Wkh, (size_t)Dc * Dc);
  cvt(Wmlp, Wmh, (size_t)HIDc * Dc);

  // 2) Q = x Wq^T ; K = x Wk^T   (bf16 out)
  dim3 gQK(Mrows / 128, Dc / 64);
  gemm_bf16_kernel<3, true><<<gQK, 256, 0, stream>>>(xh, Wqh, nullptr, Qh, Mrows, Dc, Dc);
  gemm_bf16_kernel<3, true><<<gQK, 256, 0, stream>>>(xh, Wkh, nullptr, Kh, Mrows, Dc, Dc);

  // 3) attention, two passes
  dim3 ga(Nc / 64, Bc * Hc);
  attn_pass1_kernel<<<ga, 128, 0, stream>>>(Qh, Kh, betas, AVb, mbuf, lbuf);
  attn_pass2_kernel<<<ga, 128, 0, stream>>>(Qh, Kh, betas, mbuf, lbuf, AQb);

  // 4) MLP: hid = relu(x Wmlp^T) ; out = hid Wmlp
  dim3 gH(Mrows / 128, HIDc / 64);
  gemm_bf16_kernel<2, true><<<gH, 256, 0, stream>>>(xh, Wmh, nullptr, hidb, Mrows, HIDc, Dc);
  dim3 gO(Mrows / 128, Dc / 64);
  gemm_bf16_kernel<0, false><<<gO, 256, 0, stream>>>(hidb, Wmh, out, nullptr, Mrows, Dc, HIDc);

  // 5) out += AV @ Wq ; out += AQ @ Wk
  gemm_bf16_kernel<1, false><<<gO, 256, 0, stream>>>(AVb, Wqh, out, nullptr, Mrows, Dc, Dc);
  gemm_bf16_kernel<1, false><<<gO, 256, 0, stream>>>(AQb, Wkh, out, nullptr, Mrows, Dc, Dc);
}
